// ParallelPalmTransformerLayer_9079560863769
// MI455X (gfx1250) — matrix operand order repaired and checked by reading, not compiled, not run on hardware
//
#include <hip/hip_runtime.h>

#define DIMV      2048
#define S_LEN     2048
#define BATCH     2
#define N_HEADS   32
#define D_HEAD    64
#define FFN_IN    8192
#define LDH       10368   /* q(2048)+k(64)+v(64)+ffn(8192) */
#define OUT_IN    10240
#define QK_SCALE  0.125f  /* 64^-0.5 */

typedef unsigned short u16;
typedef __attribute__((ext_vector_type(8)))  u16    v8u;
typedef __attribute__((ext_vector_type(16))) __bf16 v16bf;
typedef __attribute__((ext_vector_type(8)))  __bf16 v8bf;
typedef __attribute__((ext_vector_type(8)))  float  v8f;
typedef __attribute__((ext_vector_type(4)))  unsigned int v4u32;
typedef __attribute__((ext_vector_type(8)))  int    v8i32;
typedef __attribute__((ext_vector_type(4)))  int    v4i32;

union Frag { v16bf v; v8bf h[2]; };

__device__ __forceinline__ u16 f2bf(float f) {
  unsigned u = __float_as_uint(f);
  u += 0x7fffu + ((u >> 16) & 1u);           // round-to-nearest-even
  return (u16)(u >> 16);
}
__device__ __forceinline__ float bf2f(u16 h) {
  return __uint_as_float(((unsigned)h) << 16);
}

// --------------------------------------------------------- TDM 2-D tile load
// Loads a rows x cols (bf16) tile whose first element is at gptr, row stride
// row_stride elements, into LDS byte offset lds_off. D# layout per
// CDNA5 ISA (count=1, type=2, data_size=2B,
// tensor dims == tile dims so OOB clipping never fires; groups 2/3 zero).
__device__ __forceinline__ void tdm_load_tile(unsigned lds_off, const u16* gptr,
                                              unsigned rows, unsigned cols,
                                              unsigned row_stride)
{
  unsigned long long ga = (unsigned long long)(const void*)gptr;
  v4u32 g0;
  g0[0] = 1u;                                            // count=1, user mode
  g0[1] = lds_off;                                       // LDS byte address
  g0[2] = (unsigned)ga;                                  // global_addr[31:0]
  g0[3] = (unsigned)((ga >> 32) & 0x01ffffffu) | (2u << 30); // addr[56:32]|type=2
  v8i32 g1;
  g1[0] = (int)(1u << 16);                               // wg_mask=0, data_size=1(2B)
  g1[1] = (int)((cols & 0xffffu) << 16);                 // tensor_dim0 lo16
  g1[2] = (int)((cols >> 16) | ((rows & 0xffffu) << 16));// dim0 hi | tensor_dim1 lo
  g1[3] = (int)((rows >> 16) | ((cols & 0xffffu) << 16));// dim1 hi | tile_dim0
  g1[4] = (int)(rows & 0xffffu);                         // tile_dim1 | tile_dim2=0
  g1[5] = (int)row_stride;                               // tensor_dim0_stride lo32
  g1[6] = 0;                                             // stride hi | dim1_stride
  g1[7] = 0;
  v4i32 gz = {0, 0, 0, 0};
#if defined(__clang_major__) && (__clang_major__ >= 23)
  v8i32 gz8 = {0, 0, 0, 0, 0, 0, 0, 0};
  __builtin_amdgcn_tensor_load_to_lds(g0, g1, gz, gz, gz8, 0);
#else
  __builtin_amdgcn_tensor_load_to_lds(g0, g1, gz, gz, 0);
#endif
}

__device__ __forceinline__ unsigned lds_offset(const void* p) {
  return (unsigned)(unsigned long long)p;   // generic LDS addr: low 32 = offset
}

// ---------------------------------------------------------------- LayerNorm
__global__ __launch_bounds__(256)
void ln_kernel(const float* __restrict__ x, const float* __restrict__ g,
               const float* __restrict__ be, float* __restrict__ xn,
               u16* __restrict__ xnb)
{
  __shared__ float s1[256], s2[256];
  const int row = blockIdx.x, tid = threadIdx.x;
  const float* xr = x + (size_t)row * DIMV;
  float a = 0.f, b = 0.f;
  for (int i = tid; i < DIMV; i += 256) { float v = xr[i]; a += v; b += v * v; }
  s1[tid] = a; s2[tid] = b; __syncthreads();
  for (int off = 128; off > 0; off >>= 1) {
    if (tid < off) { s1[tid] += s1[tid + off]; s2[tid] += s2[tid + off]; }
    __syncthreads();
  }
  const float mu  = s1[0] * (1.f / DIMV);
  const float var = s2[0] * (1.f / DIMV) - mu * mu;
  const float rs  = rsqrtf(var + 1e-5f);
  for (int i = tid; i < DIMV; i += 256) {
    float v = (xr[i] - mu) * rs * g[i] + be[i];
    xn [(size_t)row * DIMV + i] = v;
    xnb[(size_t)row * DIMV + i] = f2bf(v);
  }
}

// ----------------------------------------- fp32 -> bf16 transpose (W[N][K])
__global__ __launch_bounds__(256)
void conv_tr_kernel(const float* __restrict__ W, u16* __restrict__ Wt,
                    int R, int C, int CUse)
{
  size_t idx = (size_t)blockIdx.x * 256 + threadIdx.x;
  size_t total = (size_t)CUse * R;
  if (idx >= total) return;
  int r = (int)(idx % R);
  int c = (int)(idx / R);
  Wt[(size_t)c * R + r] = f2bf(W[(size_t)r * C + c]);
}

// ---------------------------------------------------------------- GEMM bf16
// C[M x N] = A[M x K] * Bt[N x K]^T ; A may be split at ksplit between A/A2.
// Block tile 128x128x32, 8 waves (4 in M x 2 in N), wave tile 32x64.
// Tiles staged by the Tensor Data Mover into double-buffered LDS; wave 0
// issues descriptors and tracks completion with TENSORcnt.
__global__ __launch_bounds__(256)
void gemm_bf16_wmma(const u16* __restrict__ A,  int lda,
                    const u16* __restrict__ A2, int lda2, int ksplit,
                    const u16* __restrict__ Bt, int ldb,
                    u16* __restrict__ Cb, float* __restrict__ Cf,
                    const float* __restrict__ resid, int ldc, int K)
{
  __shared__ u16 lA[2][128][32];
  __shared__ u16 lB[2][128][32];
  const int tid  = threadIdx.x;
  const int lane = tid & 31;
  const int wave = tid >> 5;
  const int half = lane >> 4, lr = lane & 15;
  const int wm = wave & 3, wn = wave >> 2;
  const int mBase = blockIdx.y * 128, nBase = blockIdx.x * 128;
  const u16* bTile = Bt + (size_t)nBase * ldb;

  v8f acc[2][4];
  #pragma unroll
  for (int i = 0; i < 2; ++i)
    #pragma unroll
    for (int j = 0; j < 4; ++j)
      #pragma unroll
      for (int r = 0; r < 8; ++r) acc[i][j][r] = 0.f;

  // prologue: DMA first k-tile pair into buffer 0
  if (wave == 0) {
    const u16* ap = (0 < ksplit) ? A + (size_t)mBase * lda
                                 : A2 + (size_t)mBase * lda2;
    unsigned as = (0 < ksplit) ? (unsigned)lda : (unsigned)lda2;
    tdm_load_tile(lds_offset(&lA[0][0][0]), ap, 128, 32, as);
    tdm_load_tile(lds_offset(&lB[0][0][0]), bTile, 128, 32, (unsigned)ldb);
    __builtin_amdgcn_s_wait_tensorcnt(0);
  }
  asm volatile("" ::: "memory");
  __syncthreads();

  int buf = 0;
  for (int kb = 0; kb < K; kb += 32) {
    const int kn = kb + 32;
    if (kn < K && wave == 0) {          // DMA next pair while we compute
      const u16* ap = (kn < ksplit) ? A + (size_t)mBase * lda + kn
                                    : A2 + (size_t)mBase * lda2 + kn;
      unsigned as = (kn < ksplit) ? (unsigned)lda : (unsigned)lda2;
      tdm_load_tile(lds_offset(&lA[buf ^ 1][0][0]), ap, 128, 32, as);
      tdm_load_tile(lds_offset(&lB[buf ^ 1][0][0]), bTile + kn, 128, 32,
                    (unsigned)ldb);
    }

    Frag af[2], bfg[4];
    #pragma unroll
    for (int mt = 0; mt < 2; ++mt) {
      int ar = wm * 32 + mt * 16 + lr;
      af[mt].h[0] = *(const v8bf*)&lA[buf][ar][half * 8];       // K=h*8..+7
      af[mt].h[1] = *(const v8bf*)&lA[buf][ar][16 + half * 8];  // K=16+h*8..+7
    }
    #pragma unroll
    for (int nt = 0; nt < 4; ++nt) {
      int br = wn * 64 + nt * 16 + lr;
      bfg[nt].h[0] = *(const v8bf*)&lB[buf][br][half * 8];      // K=h*8..+7
      bfg[nt].h[1] = *(const v8bf*)&lB[buf][br][16 + half * 8]; // K=16+h*8..+7
    }
    #pragma unroll
    for (int mt = 0; mt < 2; ++mt)
      #pragma unroll
      for (int nt = 0; nt < 4; ++nt)
        acc[mt][nt] = __builtin_amdgcn_wmma_f32_16x16x32_bf16(
            false, af[mt].v, false, bfg[nt].v, (short)0, acc[mt][nt],
            false, false);

    if (wave == 0) __builtin_amdgcn_s_wait_tensorcnt(0);
    asm volatile("" ::: "memory");
    __syncthreads();
    buf ^= 1;
  }

  #pragma unroll
  for (int mt = 0; mt < 2; ++mt)
    #pragma unroll
    for (int nt = 0; nt < 4; ++nt)
      #pragma unroll
      for (int r = 0; r < 8; ++r) {
        int row = mBase + wm * 32 + mt * 16 + r + 8 * half;
        int col = nBase + wn * 64 + nt * 16 + lr;
        float v = acc[mt][nt][r];
        if (Cf) Cf[(size_t)row * ldc + col] = v + resid[(size_t)row * ldc + col];
        else    Cb[(size_t)row * ldc + col] = f2bf(v);
      }
}

// ------------------------------------------------- rotary (+q scale), in h
__global__ __launch_bounds__(256)
void rotary_kernel(u16* __restrict__ h)
{
  const int row = blockIdx.x;
  const int s = row & (S_LEN - 1);
  u16* hp = h + (size_t)row * LDH;
  for (int e = threadIdx.x; e < 33 * 32; e += 256) {
    int vec = e >> 5, i = e & 31;               // 32 q heads + 1 k vector
    int cb = (vec < 32) ? vec * 64 : 2048;
    float inv = __powf(10000.f, -(float)(2 * i) / 64.f);
    float ang = (float)s * inv;
    float c = __cosf(ang), sn = __sinf(ang);
    float a  = bf2f(hp[cb + i]);
    float bv = bf2f(hp[cb + i + 32]);
    float oa = a * c - bv * sn;
    float ob = bv * c + a * sn;
    if (vec < 32) { oa *= QK_SCALE; ob *= QK_SCALE; }
    hp[cb + i]      = f2bf(oa);
    hp[cb + i + 32] = f2bf(ob);
  }
}

// --------------------------------------------------- causal flash attention
// block = 128 threads (4 waves), each wave owns a 16-row query strip.
__global__ __launch_bounds__(128)
void attn_kernel(const u16* __restrict__ h, u16* __restrict__ ao)
{
  __shared__ u16 Vt[64][32];      // V^T tile: [dh][j]
  __shared__ u16 Pt[4][16][32];   // per-wave P tile in A layout
  const int tid  = threadIdx.x;
  const int lane = tid & 31;
  const int wave = tid >> 5;
  const int half = lane >> 4, lr = lane & 15;
  const int bb = blockIdx.z, hh = blockIdx.y, qb = blockIdx.x * 64;

  const int qrow = qb + wave * 16 + lr;
  const u16* qp = h + (size_t)(bb * S_LEN + qrow) * LDH + hh * D_HEAD;

  Frag qf[2];
  #pragma unroll
  for (int ks = 0; ks < 2; ++ks) {
    int k0 = ks * 32 + half * 8;
    qf[ks].h[0] = *(const v8bf*)(qp + k0);
    qf[ks].h[1] = *(const v8bf*)(qp + k0 + 16);
  }

  v8f o[4];
  float m_i[8], l_i[8];
  #pragma unroll
  for (int t = 0; t < 4; ++t)
    #pragma unroll
    for (int r = 0; r < 8; ++r) o[t][r] = 0.f;
  #pragma unroll
  for (int r = 0; r < 8; ++r) { m_i[r] = -3.0e38f; l_i[r] = 0.f; }

  const int jend = qb + 64;
  for (int jb = 0; jb < jend; jb += 32) {
    if (jb + 32 < jend)                 // prefetch next K tile rows
      __builtin_prefetch(h + (size_t)(bb * S_LEN + jb + 32 + lr) * LDH + 2048,
                         0, 1);
    __syncthreads();
    { // stage V^T cooperatively: V[jb..jb+31][0..63] -> Vt[dh][j]
      int j = tid >> 2, d0 = (tid & 3) * 16;
      const u16* vp = h + (size_t)(bb * S_LEN + jb + j) * LDH + 2112 + d0;
      v8u a0 = *(const v8u*)vp, a1 = *(const v8u*)(vp + 8);
      #pragma unroll
      for (int i = 0; i < 8; ++i) {
        Vt[d0 + i][j]     = a0[i];
        Vt[d0 + 8 + i][j] = a1[i];
      }
    }
    __syncthreads();

    // S = Q K^T  (two 16-col subtiles, K rows already match B-layout)
    v8f sa[2];
    #pragma unroll
    for (int sub = 0; sub < 2; ++sub) {
      #pragma unroll
      for (int r = 0; r < 8; ++r) sa[sub][r] = 0.f;
      int jcol = jb + sub * 16 + lr;
      const u16* kp = h + (size_t)(bb * S_LEN + jcol) * LDH + 2048;
      #pragma unroll
      for (int ks = 0; ks < 2; ++ks) {
        Frag kf;
        int kk = ks * 32 + half * 8;
        kf.h[0] = *(const v8bf*)(kp + kk);
        kf.h[1] = *(const v8bf*)(kp + kk + 16);
        sa[sub] = __builtin_amdgcn_wmma_f32_16x16x32_bf16(
            false, qf[ks].v, false, kf.v, (short)0, sa[sub], false, false);
      }
    }

    // causal mask + online softmax update
    float mnew[8], alpha[8], psum[8];
    #pragma unroll
    for (int r = 0; r < 8; ++r) {
      int qr = qb + wave * 16 + r + 8 * half;
      if (jb + lr      > qr) sa[0][r] = -1.0e30f;
      if (jb + 16 + lr > qr) sa[1][r] = -1.0e30f;
      float mx = fmaxf(sa[0][r], sa[1][r]);
      #pragma unroll
      for (int msk = 1; msk < 16; msk <<= 1)
        mx = fmaxf(mx, __shfl_xor(mx, msk, 32));
      mnew[r]  = fmaxf(m_i[r], mx);
      alpha[r] = __expf(m_i[r] - mnew[r]);
      m_i[r]   = mnew[r];
      l_i[r]  *= alpha[r];
      psum[r]  = 0.f;
    }
    #pragma unroll
    for (int t = 0; t < 4; ++t)
      #pragma unroll
      for (int r = 0; r < 8; ++r) o[t][r] *= alpha[r];

    #pragma unroll
    for (int sub = 0; sub < 2; ++sub)
      #pragma unroll
      for (int r = 0; r < 8; ++r) {
        float p = __expf(sa[sub][r] - mnew[r]);
        psum[r] += p;
        Pt[wave][r + 8 * half][sub * 16 + lr] = f2bf(p);
      }
    #pragma unroll
    for (int r = 0; r < 8; ++r) {
      float sv = psum[r];
      #pragma unroll
      for (int msk = 1; msk < 16; msk <<= 1) sv += __shfl_xor(sv, msk, 32);
      l_i[r] += sv;
    }

    asm volatile("" ::: "memory");  // keep same-wave LDS write->read order

    // O += P V  (P as A-frag from LDS, V^T rows are B-layout contiguous)
    Frag pf;
    pf.h[0] = *(const v8bf*)&Pt[wave][lr][half * 8];
    pf.h[1] = *(const v8bf*)&Pt[wave][lr][16 + half * 8];
    #pragma unroll
    for (int dt = 0; dt < 4; ++dt) {
      Frag vf;
      vf.h[0] = *(const v8bf*)&Vt[dt * 16 + lr][half * 8];
      vf.h[1] = *(const v8bf*)&Vt[dt * 16 + lr][16 + half * 8];
      o[dt] = __builtin_amdgcn_wmma_f32_16x16x32_bf16(
          false, pf.v, false, vf.v, (short)0, o[dt], false, false);
    }
  }

  #pragma unroll
  for (int dt = 0; dt < 4; ++dt)
    #pragma unroll
    for (int r = 0; r < 8; ++r) {
      int row = bb * S_LEN + qb + wave * 16 + r + 8 * half;
      int col = hh * D_HEAD + dt * 16 + lr;
      ao[(size_t)row * 2048 + col] = f2bf(o[dt][r] / l_i[r]);
    }
}

// --------------------------------------------------------------- launcher
extern "C" void kernel_launch(void* const* d_in, const int* in_sizes, int n_in,
                              void* d_out, int out_size, void* d_ws, size_t ws_size,
                              hipStream_t stream)
{
  (void)in_sizes; (void)n_in; (void)out_size; (void)ws_size;
  const float* x  = (const float*)d_in[0];
  const float* Wi = (const float*)d_in[1];
  const float* Wo = (const float*)d_in[2];
  const float* g  = (const float*)d_in[3];
  const float* be = (const float*)d_in[4];
  float* out = (float*)d_out;

  const size_t ROWS = (size_t)BATCH * S_LEN;   // 4096
  char* ws = (char*)d_ws;
  float* xn_f   = (float*)ws;  ws += ROWS * DIMV * 4;
  u16*   xn_b   = (u16*)ws;    ws += ROWS * DIMV * 2;
  u16*   Wt_in  = (u16*)ws;    ws += (size_t)LDH * DIMV * 2;     // [10368][2048]
  u16*   Wt_out = (u16*)ws;    ws += (size_t)DIMV * OUT_IN * 2;  // [2048][10240]
  u16*   hbuf   = (u16*)ws;    ws += ROWS * LDH * 2;
  u16*   attn_o = (u16*)ws;

  ln_kernel<<<(int)ROWS, 256, 0, stream>>>(x, g, be, xn_f, xn_b);

  conv_tr_kernel<<<(int)(((size_t)LDH * DIMV + 255) / 256), 256, 0, stream>>>(
      Wi, Wt_in, DIMV, 18560, LDH);
  conv_tr_kernel<<<(int)(((size_t)DIMV * OUT_IN + 255) / 256), 256, 0, stream>>>(
      Wo, Wt_out, OUT_IN, DIMV, DIMV);

  // h = xn @ W_in[:, :10368]     (M=4096, N=10368, K=2048)
  gemm_bf16_wmma<<<dim3(LDH / 128, (int)ROWS / 128), 256, 0, stream>>>(
      xn_b, DIMV, xn_b, DIMV, DIMV, Wt_in, DIMV,
      hbuf, nullptr, nullptr, LDH, DIMV);

  rotary_kernel<<<(int)ROWS, 256, 0, stream>>>(hbuf);

  attn_kernel<<<dim3(S_LEN / 64, N_HEADS, BATCH), 128, 0, stream>>>(hbuf, attn_o);

  // out = [attn_out | ffn] @ W_out + xn   (M=4096, N=2048, K=10240; split at 2048)
  gemm_bf16_wmma<<<dim3(DIMV / 128, (int)ROWS / 128), 256, 0, stream>>>(
      attn_o, DIMV, hbuf + 128, LDH, DIMV, Wt_out, OUT_IN,
      nullptr, out, xn_f, DIMV, OUT_IN);
}
